// ContrastiveLoss_66503273611694
// MI455X (gfx1250) — matrix operand order repaired and checked by reading, not compiled, not run on hardware
//
#include <hip/hip_runtime.h>
#include <hip/hip_bf16.h>

// ---------------------------------------------------------------------------
// ContrastiveLoss on MI455X (gfx1250): fused WMMA similarity + masked reduce.
// Never materializes the 4000x4000 sim matrix (would be 512MB of HBM traffic);
// computes 16x16 f32 tiles in VGPRs via v_wmma_f32_16x16x32_f16 and folds the
// exp / softplus mask reductions straight out of the accumulator registers
// using raw v_exp_f32 / v_log_f32 / v_rcp_f32 (branchless, cndmask-selected).
// Panel staging uses gfx1250 async global->LDS DMA (ASYNCcnt-tracked).
// ---------------------------------------------------------------------------

typedef __attribute__((ext_vector_type(16))) _Float16     v16h;
typedef __attribute__((ext_vector_type(8)))  float        v8f;
typedef __attribute__((ext_vector_type(4)))  unsigned int u32x4;
typedef __attribute__((ext_vector_type(4)))  int          v4i;

typedef __attribute__((address_space(1))) v4i gas_v4i;   // global int4
typedef __attribute__((address_space(3))) v4i las_v4i;   // LDS int4

union FragU {
    struct { u32x4 lo, hi; } q;   // 32 bytes
    v16h h;
};

#define BS   8
#define CH   64
#define NPT  4000
#define SUP  80          // super-tile rows per block (5 x 16)
#define NT   5           // 16-row tiles per super-tile
#define NBLK 50          // 4000 / 80

// exp(sim/T) = exp2( c * (1/T) * log2(e) ) ; ln(x) = log2(x) * ln(2)
#define EXP_SCALE 14.426950408889634f   // 10 * log2(e)
#define LN2       0.6931471805599453f

// 16B async global->LDS copy (GLOBAL_LOAD_ASYNC_TO_LDS_B128, ASYNCcnt).
__device__ static inline void async_cp16(const void* g, void* l) {
    __builtin_amdgcn_global_load_async_to_lds_b128(
        (gas_v4i*)(void*)g, (las_v4i*)l, 0, 0);
}

// ---------------------------------------------------------------- init ------
__global__ void cl_init(int* cnt, float* pos_sum, float* neg_sum) {
    int t = threadIdx.x;
    if (t < BS) { cnt[t] = 0; pos_sum[t] = 0.0f; neg_sum[t] = 0.0f; }
}

// ----------------------------------------------------------- normalize ------
// features layout: (BS, C, N) -> element (b,c,n) at feat[(b*CH + c)*NPT + n].
// Output featn: row-major (b*NPT + n)*CH + c as f16 (contiguous 128B rows).
__global__ void cl_normalize(const float* __restrict__ feat,
                             const float* __restrict__ prob,
                             _Float16* __restrict__ featn,
                             unsigned char* __restrict__ nm,
                             int* __restrict__ cnt) {
    __shared__ float red[256];
    int tid  = threadIdx.x;
    int rloc = tid >> 6;            // 4 rows per 256-thread block
    int c    = tid & 63;
    int row  = blockIdx.x * 4 + rloc;   // 0 .. BS*NPT-1
    int b    = row / NPT;
    int n    = row - b * NPT;

    float x = feat[(b * CH + c) * NPT + n];
    red[tid] = x * x;
    __syncthreads();
    for (int s = 32; s > 0; s >>= 1) {
        if (c < s) red[tid] += red[tid + s];
        __syncthreads();
    }
    float ss    = red[rloc << 6];
    float scale = 1.0f / fmaxf(sqrtf(ss), 1e-12f);
    featn[(size_t)row * CH + c] = (_Float16)(x * scale);

    if (c == 0) {
        bool normal = prob[row] < 0.5f;     // prob index (b*NPT + n) == row
        nm[row] = normal ? 1 : 0;
        if (normal) atomicAdd(&cnt[b], 1);
    }
}

// ------------------------------------------------------------ main pass -----
__global__ void __launch_bounds__(256)
cl_contrast(const _Float16* __restrict__ featn,
            const unsigned char* __restrict__ nm,
            float* __restrict__ pos_sum,
            float* __restrict__ neg_sum) {
    __shared__ __align__(16) _Float16 sA[SUP * CH];
    __shared__ __align__(16) _Float16 sB[SUP * CH];
    __shared__ float redp[256];
    __shared__ float redn[256];

    const int tid = threadIdx.x;
    const int b   = blockIdx.z;
    const int i0  = blockIdx.y * SUP;
    const int j0  = blockIdx.x * SUP;

    // ---- stage both 80-row panels (10KB each, contiguous in global) -------
    // via async global->LDS DMA: no VGPR round-trip, tracked by ASYNCcnt.
    const u32x4* gA = (const u32x4*)(featn + (size_t)(b * NPT + i0) * CH);
    const u32x4* gB = (const u32x4*)(featn + (size_t)(b * NPT + j0) * CH);
    u32x4* lA = (u32x4*)sA;
    u32x4* lB = (u32x4*)sB;
    const int NV = (SUP * CH * (int)sizeof(_Float16)) / 16;   // 640
    for (int k = tid; k < NV; k += 256) {
        async_cp16(&gA[k], &lA[k]);
        async_cp16(&gB[k], &lB[k]);
    }
    asm volatile("s_wait_asynccnt 0x0" ::: "memory");
    __syncthreads();

    const int wave = tid >> 5;
    const int lane = tid & 31;
    const int half = lane >> 4;     // 0: lanes 0-15, 1: lanes 16-31
    const int l16  = lane & 15;

    const char* sAb = (const char*)sA;
    const char* sBb = (const char*)sB;

    float pos_acc = 0.0f, neg_acc = 0.0f;

    // 25 tile-pairs per block, round-robined over 8 waves (uniform per wave,
    // so EXEC stays all-ones around the WMMAs).
    for (int t = wave; t < NT * NT; t += 8) {
        const int ti = t / NT;
        const int tj = t - ti * NT;

        // ---- A fragments (ISA 16-bit A layout: lane=M, K split per half) --
        // kc=0: elems 0-7 = K(half?8:0)..+7, elems 8-15 = K(half?24:16)..+7
        const int arow = ti * 16 + l16;
        const int ab   = arow * 128 + half * 16;
        FragU a0, a1;
        a0.q.lo = *(const u32x4*)(sAb + ab);
        a0.q.hi = *(const u32x4*)(sAb + ab + 32);
        a1.q.lo = *(const u32x4*)(sAb + ab + 64);
        a1.q.hi = *(const u32x4*)(sAb + ab + 96);

        // ---- B fragments (lane=N; same K split per half as A) -------------
        const int brow = tj * 16 + l16;
        const int bb   = brow * 128 + half * 16;
        FragU b0, b1;
        b0.q.lo = *(const u32x4*)(sBb + bb);
        b0.q.hi = *(const u32x4*)(sBb + bb + 32);
        b1.q.lo = *(const u32x4*)(sBb + bb + 64);
        b1.q.hi = *(const u32x4*)(sBb + bb + 96);

        // ---- sim tile: K=64 via two chained WMMAs -------------------------
        v8f c = {};
        c = __builtin_amdgcn_wmma_f32_16x16x32_f16(false, a0.h, false, b0.h,
                                                   (short)0, c, false, false);
        c = __builtin_amdgcn_wmma_f32_16x16x32_f16(false, a1.h, false, b1.h,
                                                   (short)0, c, false, false);

        // ---- masked epilogue straight out of accumulator VGPRs ------------
        // C/D layout: lane (n + 16*(m>=8)) vgpr m%8 holds element (m, n).
        const int gi_base = i0 + ti * 16 + half * 8;
        const int gj      = j0 + tj * 16 + l16;
        const bool jn     = nm[(size_t)b * NPT + gj] != 0;
        const unsigned long long mrow =
            *(const unsigned long long*)(nm + (size_t)b * NPT + gi_base);

#pragma unroll
        for (int k = 0; k < 8; ++k) {
            const bool  im = ((mrow >> (8 * k)) & 0xFFull) != 0;
            const int   gi = gi_base + k;
            // e = exp(sim) = exp2(c * 10*log2(e))   -> single v_exp_f32
            const float e  = __builtin_amdgcn_exp2f(c[k] * EXP_SCALE);
            // 1 - sigmoid(s) = 1/(1+e)              -> single v_rcp_f32
            const float om = __builtin_amdgcn_rcpf(1.0f + e);
            // -ln(om + eps) = -ln2 * log2(om + eps) -> single v_log_f32
            const float nl = -LN2 * __builtin_amdgcn_logf(om + 1e-6f);
            const float pm = (im && jn && (gi != gj)) ? 1.0f : 0.0f;
            const float nq = (im && !jn)              ? 1.0f : 0.0f;
            pos_acc += pm * e;
            neg_acc += nq * nl;
        }
    }

    // ---- block reduction -> one atomic per batch accumulator --------------
    redp[tid] = pos_acc;
    redn[tid] = neg_acc;
    __syncthreads();
    for (int s = 128; s > 0; s >>= 1) {
        if (tid < s) { redp[tid] += redp[tid + s]; redn[tid] += redn[tid + s]; }
        __syncthreads();
    }
    if (tid == 0) {
        atomicAdd(&pos_sum[b], redp[0]);
        atomicAdd(&neg_sum[b], redn[0]);
    }
}

// ------------------------------------------------------------- finalize -----
__global__ void cl_finalize(const int* __restrict__ cnt,
                            const float* __restrict__ pos_sum,
                            const float* __restrict__ neg_sum,
                            float* __restrict__ out) {
    if (threadIdx.x != 0) return;
    float total = 0.0f, nvalid = 0.0f;
    for (int b = 0; b < BS; ++b) {
        int   ni = cnt[b];
        float nn = (float)ni;
        float na = (float)(NPT - ni);
        bool  valid = (ni >= 10) && ((NPT - ni) >= 5);
        float pos_cnt  = nn * nn - nn;                       // n^2 - n
        float pos_mean = pos_sum[b] / fmaxf(pos_cnt, 1.0f);
        float pl = -logf(pos_mean + 1e-6f);
        float cross = nn * na;
        float nl = neg_sum[b] / fmaxf(cross, 1.0f);
        if (valid) { total += pl + nl; nvalid += 1.0f; }
    }
    out[0] = total / fmaxf(nvalid, 1.0f);
}

// -------------------------------------------------------------- launch ------
extern "C" void kernel_launch(void* const* d_in, const int* in_sizes, int n_in,
                              void* d_out, int out_size, void* d_ws, size_t ws_size,
                              hipStream_t stream) {
    (void)in_sizes; (void)n_in; (void)out_size; (void)ws_size;

    const float* features = (const float*)d_in[0];   // 8*64*4000
    const float* prob     = (const float*)d_in[1];   // 8*4000

    // workspace layout
    char*          ws      = (char*)d_ws;
    int*           cnt     = (int*)ws;                      // [8]
    float*         pos_sum = (float*)(ws + 32);             // [8]
    float*         neg_sum = (float*)(ws + 64);             // [8]
    unsigned char* nmask   = (unsigned char*)(ws + 128);    // [8*4000]
    _Float16*      featn   = (_Float16*)(ws + 32768);       // [8*4000*64]

    float* out = (float*)d_out;

    cl_init<<<1, 32, 0, stream>>>(cnt, pos_sum, neg_sum);
    cl_normalize<<<(BS * NPT) / 4, 256, 0, stream>>>(features, prob,
                                                     featn, nmask, cnt);
    dim3 grid(NBLK, NBLK, BS);
    cl_contrast<<<grid, 256, 0, stream>>>(featn, nmask, pos_sum, neg_sum);
    cl_finalize<<<1, 32, 0, stream>>>(cnt, pos_sum, neg_sum, out);
}
